// SRGNN_60421599920761
// MI455X (gfx1250) — matrix operand order repaired and checked by reading, not compiled, not run on hardware
//
#include <hip/hip_runtime.h>
#include <hip/hip_bf16.h>

typedef _Float16 h16;
typedef __attribute__((ext_vector_type(16))) _Float16 v16h;
typedef __attribute__((ext_vector_type(8)))  _Float16 v8h;
typedef __attribute__((ext_vector_type(8)))  float    v8f;

#define DEV static __device__ __forceinline__

// ---------------------------------------------------------------------------
// CDNA5 async global->LDS copy (ASYNCcnt-tracked), 16B per lane per issue.
// LDS operand = low 32 bits of the shared-space address (flat->LDS mapping
// truncates to addr[31:0]; kernel uses only dynamic LDS, so offset == address).
DEV void async_ld_b128(void* lds_ptr, const void* gptr) {
  const unsigned l = (unsigned)(unsigned long long)lds_ptr;
  asm volatile("global_load_async_to_lds_b128 %0, %1, off"
               :: "v"(l), "v"(gptr) : "memory");
}
DEV void wait_async0() { asm volatile("s_wait_asynccnt 0" ::: "memory"); }

// ---------------------------------------------------------------------------
// WMMA helpers (CDNA5 wave32, V_WMMA_F32_16X16X32_F16)
//
// A operand: 16x32 (MxK) f16 tile, row-major source with leading dim `ld`.
// Lane L holds row M=L%16; VGPR0-7 = K[hf*16..+15].
DEV v16h load_a16(const h16* base, int ld) {
  const int lane = threadIdx.x & 31;
  const int r = lane & 15, hf = lane >> 4;
  const h16* p = base + r * ld + hf * 16;
  union { v16h v; v8h h[2]; } u;
  u.h[0] = *(const v8h*)(p);
  u.h[1] = *(const v8h*)(p + 8);
  return u.v;
}

// B operand for D = X * W^T: B[k,n'] = W[n',k], W row-major [N x K].
// Lane L holds col n'=L%16, K[hf*16..+15] -> one contiguous 32B load.
DEV v16h load_bT(const h16* base, int ld) {
  const int lane = threadIdx.x & 31;
  const int r = lane & 15, hf = lane >> 4;
  return *(const v16h*)(base + r * ld + hf * 16);
}

DEV v8f wmma_f16(v16h a, v16h b, v8f c) {
  return __builtin_amdgcn_wmma_f32_16x16x32_f16(
      false, a, false, b, (short)0, c, false, false);
}

DEV float sigmoidf_(float x) { return 1.0f / (1.0f + __expf(-x)); }

// ---------------------------------------------------------------------------
// fp32 -> f16 conversion (weights and A)
__global__ void cvt_f16_kernel(const float* __restrict__ src,
                               h16* __restrict__ dst, int n) {
  for (int i = blockIdx.x * blockDim.x + threadIdx.x; i < n;
       i += gridDim.x * blockDim.x)
    dst[i] = (h16)src[i];
}

// hidden = emb[items]  (f16), 65536 rows x 128
__global__ void gather_emb_kernel(const float* __restrict__ emb,
                                  const int* __restrict__ items,
                                  h16* __restrict__ hidden_h) {
  const int i = blockIdx.x * 256 + threadIdx.x;  // exact grid: 32768 blocks
  const int row = i >> 7, col = i & 127;
  hidden_h[i] = (h16)emb[(size_t)items[row] * 128 + col];
}

// ---------------------------------------------------------------------------
// One fused GRU step: one workgroup per batch element b.
__global__ __launch_bounds__(256)
void gru_step_kernel(const h16* __restrict__ A_h,      // [B,64,128] f16
                     const float* __restrict__ b_in,  const float* __restrict__ b_out,
                     const float* __restrict__ b_iah, const float* __restrict__ b_oah,
                     const float* __restrict__ bi,    const float* __restrict__ bh,
                     const h16*  __restrict__ w_in_h, const h16* __restrict__ w_out_h,
                     const h16*  __restrict__ wi_h,   const h16* __restrict__ wh_h,
                     h16* __restrict__ hidden_h)       // [B,64,128] in/out
{
  extern __shared__ __align__(32) h16 smem[];
  h16* Hs    = smem;             // [64 x 128] hidden (f16)
  h16* HinT  = Hs    + 64*128;   // [128 x 64] h_in^T
  h16* HoutT = HinT  + 128*64;   // [128 x 64] h_out^T
  h16* As    = HoutT + 128*64;   // [64 x 128] A (f16): cols 0-63 in, 64-127 out
  h16* IE    = As    + 64*128;   // [64 x 256] input_emb

  const int b    = blockIdx.x;
  const int tid  = threadIdx.x;
  const int wave = tid >> 5;
  const int lane = tid & 31;
  const int r16  = lane & 15, hf = lane >> 4;

  // ---- async-stage hidden + A into LDS (16B/lane/issue, ASYNCcnt) ---------
  const h16* hsrc = hidden_h + (size_t)b * 8192;
  const h16* asrc = A_h      + (size_t)b * 8192;
#pragma unroll
  for (int k = 0; k < 4; ++k) {             // 4*256 = 1024 x 16B = 16KB each
    const int i = tid + k * 256;
    async_ld_b128(Hs + i * 8, hsrc + i * 8);
    async_ld_b128(As + i * 8, asrc + i * 8);
  }
  // warm WGP$ with streamed weights while the DMA runs
  __builtin_prefetch(w_in_h  + tid * 64, 0, 3);
  __builtin_prefetch(wi_h    + tid * 384, 0, 3);
  __builtin_prefetch(wh_h    + tid * 192, 0, 3);
  wait_async0();
  __syncthreads();

  // ---- stage 1: h_in / h_out  (64 rows x 128 cols each) -------------------
  for (int t = wave; t < 64; t += 8) {
    const int tt = t & 31;
    const int mo = (tt >> 3) << 4;
    const int jo = (tt & 7) << 4;
    const h16*   W    = (t < 32) ? w_in_h : w_out_h;
    const float* bias = (t < 32) ? b_in   : b_out;
    h16*         dstT = (t < 32) ? HinT   : HoutT;
    v8f acc = {};
#pragma unroll
    for (int kb = 0; kb < 128; kb += 32) {
      v16h a  = load_a16(Hs + mo * 128 + kb, 128);
      v16h bm = load_bT(W  + jo * 128 + kb, 128);
      acc = wmma_f16(a, bm, acc);
    }
    const float bj = bias[jo + r16];
    v8h pk;
#pragma unroll
    for (int r = 0; r < 8; ++r) pk[r] = (h16)(acc[r] + bj);
    // transposed store: row j = jo+r16, cols m = mo + hf*8 + (0..7)
    *(v8h*)(dstT + (jo + r16) * 64 + mo + hf * 8) = pk;
  }
  __syncthreads();

  // ---- stage 2: input_in / input_out  (K = 64) ----------------------------
  for (int t = wave; t < 64; t += 8) {
    const int tt = t & 31;
    const int mo = (tt >> 3) << 4;
    const int jo = (tt & 7) << 4;
    const bool   inHalf = (t < 32);
    const h16*   Bt   = inHalf ? HinT : HoutT;
    const float* bias = inHalf ? b_iah : b_oah;
    const int    acol = inHalf ? 0 : 64;
    const int    ocol = inHalf ? 0 : 128;
    v8f acc = {};
#pragma unroll
    for (int kb = 0; kb < 64; kb += 32) {
      v16h a  = load_a16(As + mo * 128 + acol + kb, 128);
      v16h bm = load_bT(Bt + jo * 64 + kb, 64);   // B[k,n'] = h_inT[n',k]
      acc = wmma_f16(a, bm, acc);
    }
    const float bj = bias[jo + r16];
#pragma unroll
    for (int r = 0; r < 8; ++r)
      IE[(mo + r + hf * 8) * 256 + ocol + jo + r16] = (h16)(acc[r] + bj);
  }
  __syncthreads();

  // ---- stage 3: six gate GEMM tiles + GRU gate math, fused ----------------
  for (int t = wave; t < 32; t += 8) {
    const int mo = (t >> 3) << 4;
    const int jo = (t & 7) << 4;
    v8f a_ir = {}, a_ii = {}, a_in = {};
    v8f a_hr = {}, a_hi = {}, a_hn = {};
#pragma unroll
    for (int kb = 0; kb < 256; kb += 32) {   // gi = IE * wi^T, K = 256
      v16h a = load_a16(IE + mo * 256 + kb, 256);
      a_ir = wmma_f16(a, load_bT(wi_h + (jo)       * 256 + kb, 256), a_ir);
      a_ii = wmma_f16(a, load_bT(wi_h + (128 + jo) * 256 + kb, 256), a_ii);
      a_in = wmma_f16(a, load_bT(wi_h + (256 + jo) * 256 + kb, 256), a_in);
    }
#pragma unroll
    for (int kb = 0; kb < 128; kb += 32) {   // gh = H * wh^T, K = 128
      v16h a = load_a16(Hs + mo * 128 + kb, 128);
      a_hr = wmma_f16(a, load_bT(wh_h + (jo)       * 128 + kb, 128), a_hr);
      a_hi = wmma_f16(a, load_bT(wh_h + (128 + jo) * 128 + kb, 128), a_hi);
      a_hn = wmma_f16(a, load_bT(wh_h + (256 + jo) * 128 + kb, 128), a_hn);
    }
    const int   j   = jo + r16;
    const float bir = bi[j], bii = bi[128 + j], bin_ = bi[256 + j];
    const float bhr = bh[j], bhi = bh[128 + j], bhn  = bh[256 + j];
    h16* hdst = hidden_h + (size_t)b * 8192;
#pragma unroll
    for (int r = 0; r < 8; ++r) {
      const int   m  = mo + r + hf * 8;
      const float rg = sigmoidf_((a_ir[r] + bir) + (a_hr[r] + bhr)); // reset
      const float zg = sigmoidf_((a_ii[r] + bii) + (a_hi[r] + bhi)); // update
      const float ng = tanhf(rg * (a_hn[r] + bhn) + (a_in[r] + bin_));
      const float ho = (float)Hs[m * 128 + j];
      hdst[m * 128 + j] = (h16)((1.0f - zg) * ho + zg * ng);
    }
  }
}

// ---------------------------------------------------------------------------
// Attention / readout: one workgroup per batch element.
__global__ __launch_bounds__(256)
void attn_kernel(const h16* __restrict__ hidden_h,
                 const int* __restrict__ alias_in, const int* __restrict__ mask,
                 const float* __restrict__ w1, const float* __restrict__ b1,
                 const h16*  __restrict__ w2_h, const float* __restrict__ b2,
                 const float* __restrict__ wq,
                 const float* __restrict__ w3, const float* __restrict__ b3,
                 float* __restrict__ out)
{
  extern __shared__ __align__(32) h16 smem2[];
  h16*   SH      = smem2;                         // [64 x 128] seq_hidden
  float* le      = (float*)(SH + 64 * 128);       // [128]
  float* q1s     = le + 128;                      // [128]
  float* ges     = q1s + 128;                     // [128]
  float* alpha_s = ges + 128;                     // [64]
  float* am      = alpha_s + 64;                  // [64]
  int*   s_lastp = (int*)(am + 64);

  const int b    = blockIdx.x;
  const int tid  = threadIdx.x;
  const int wave = tid >> 5;
  const int lane = tid & 31;
  const int r16  = lane & 15, hf = lane >> 4;

  if (tid == 0) {
    int s = 0;
    for (int l = 0; l < 64; ++l) s += mask[b * 64 + l];
    *s_lastp = s - 1;
  }
  if (tid < 64) alpha_s[tid] = 0.0f;
  // async gather of seq_hidden rows by alias_input (per-lane addresses)
  const h16* hb = hidden_h + (size_t)b * 8192;
#pragma unroll
  for (int k = 0; k < 4; ++k) {            // 1024 x 16B chunks
    const int i = tid + k * 256;           // chunk index
    const int l = i >> 4, c = i & 15;      // row l, 16B-chunk c within row
    async_ld_b128(SH + i * 8, hb + alias_in[b * 64 + l] * 128 + c * 8);
  }
  wait_async0();
  __syncthreads();

  const int s_last = *s_lastp;
  if (tid < 128) le[tid] = (float)SH[s_last * 128 + tid];   // local_emb
  __syncthreads();

  if (tid < 128) {                                 // q1 = local_emb*w1^T + b1
    float s = b1[tid];
    const float* wr = w1 + tid * 128;
    for (int k = 0; k < 128; ++k) s += le[k] * wr[k];
    q1s[tid] = s;
  }
  __syncthreads();

  // q2 = SH * w2^T + b2 (WMMA) fused with alpha = sigmoid(q1+q2) @ wq^T
  for (int t = wave; t < 32; t += 8) {
    const int mo = (t >> 3) << 4;
    const int jo = (t & 7) << 4;
    v8f acc = {};
#pragma unroll
    for (int kb = 0; kb < 128; kb += 32) {
      v16h a  = load_a16(SH + mo * 128 + kb, 128);
      v16h bm = load_bT(w2_h + jo * 128 + kb, 128);
      acc = wmma_f16(a, bm, acc);
    }
    const int   j   = jo + r16;
    const float q1v = q1s[j], b2v = b2[j], wqv = wq[j];
#pragma unroll
    for (int r = 0; r < 8; ++r) {
      const float contrib = sigmoidf_(q1v + acc[r] + b2v) * wqv;
      atomicAdd(&alpha_s[mo + r + hf * 8], contrib);     // ds_add_f32
    }
  }
  __syncthreads();

  if (tid < 64) am[tid] = alpha_s[tid] * (float)mask[b * 64 + tid];
  __syncthreads();

  if (tid < 128) {                                 // global_emb
    float s = 0.0f;
    for (int l = 0; l < 64; ++l) s += am[l] * (float)SH[l * 128 + tid];
    ges[tid] = s;
  }
  __syncthreads();

  if (tid < 128) {                                 // out = [le|ge]*w3^T + b3
    const float* wr = w3 + tid * 256;
    float s = b3[tid];
    for (int j = 0; j < 128; ++j) s += le[j] * wr[j];
    for (int j = 0; j < 128; ++j) s += ges[j] * wr[128 + j];
    out[b * 128 + tid] = s;
  }
}

// ---------------------------------------------------------------------------
extern "C" void kernel_launch(void* const* d_in, const int* in_sizes, int n_in,
                              void* d_out, int out_size, void* d_ws, size_t ws_size,
                              hipStream_t stream) {
  (void)in_sizes; (void)n_in; (void)out_size; (void)ws_size;
  const float* A     = (const float*)d_in[0];
  const int*   items = (const int*)d_in[1];
  const int*   mask  = (const int*)d_in[2];
  const int*   alias = (const int*)d_in[3];
  const float* emb   = (const float*)d_in[4];
  const float* w_in  = (const float*)d_in[5];
  const float* b_in  = (const float*)d_in[6];
  const float* w_out = (const float*)d_in[7];
  const float* b_out = (const float*)d_in[8];
  const float* b_iah = (const float*)d_in[9];
  const float* b_oah = (const float*)d_in[10];
  const float* wi    = (const float*)d_in[11];
  const float* bi    = (const float*)d_in[12];
  const float* wh    = (const float*)d_in[13];
  const float* bh    = (const float*)d_in[14];
  const float* w1    = (const float*)d_in[15];
  const float* b1    = (const float*)d_in[16];
  const float* w2    = (const float*)d_in[17];
  const float* b2    = (const float*)d_in[18];
  const float* wq    = (const float*)d_in[19];
  const float* w3    = (const float*)d_in[20];
  const float* b3    = (const float*)d_in[21];
  float* out = (float*)d_out;

  // workspace carve-out (f16): hidden + A + weight copies (~33 MB)
  h16* p = (h16*)d_ws;
  h16* hidden_h = p; p += (size_t)1024 * 64 * 128;
  h16* A_h     = p; p += (size_t)1024 * 64 * 128;
  h16* w_in_h  = p; p += 128 * 128;
  h16* w_out_h = p; p += 128 * 128;
  h16* wi_h    = p; p += 384 * 256;
  h16* wh_h    = p; p += 384 * 128;
  h16* w2_h    = p; p += 128 * 128;

  // precision conversion (f32 -> f16): A once, weights once
  cvt_f16_kernel<<<8192, 256, 0, stream>>>(A,     A_h,     1024 * 64 * 128);
  cvt_f16_kernel<<<64,  256, 0, stream>>>(w_in,  w_in_h,  128 * 128);
  cvt_f16_kernel<<<64,  256, 0, stream>>>(w_out, w_out_h, 128 * 128);
  cvt_f16_kernel<<<384, 256, 0, stream>>>(wi,    wi_h,    384 * 256);
  cvt_f16_kernel<<<192, 256, 0, stream>>>(wh,    wh_h,    384 * 128);
  cvt_f16_kernel<<<64,  256, 0, stream>>>(w2,    w2_h,    128 * 128);

  // hidden = emb[items]
  gather_emb_kernel<<<32768, 256, 0, stream>>>(emb, items, hidden_h);

  // two fused GRU steps, 96 KB dynamic LDS per workgroup
  const size_t lds = (size_t)(64*128 + 128*64 + 128*64 + 64*128 + 64*256) * sizeof(h16);
  for (int step = 0; step < 2; ++step)
    gru_step_kernel<<<1024, 256, lds, stream>>>(
        A_h, b_in, b_out, b_iah, b_oah, bi, bh,
        w_in_h, w_out_h, wi_h, wh_h, hidden_h);

  // attention + readout (SH + reduction scratch in dynamic LDS)
  const size_t lds2 = (size_t)(64 * 128) * sizeof(h16)
                    + (128 * 3 + 64 * 2 + 4) * sizeof(float);
  attn_kernel<<<1024, 256, lds2, stream>>>(hidden_h, alias, mask,
                                           w1, b1, w2_h, b2, wq, w3, b3, out);
}
